// Encoder_30253749633670
// MI455X (gfx1250) — matrix operand order repaired and checked by reading, not compiled, not run on hardware
//
#include <hip/hip_runtime.h>
#include <hip/hip_bf16.h>
#include <math.h>

typedef __attribute__((ext_vector_type(16))) _Float16 v16h;
typedef __attribute__((ext_vector_type(8)))  float    v8f;
typedef int g_v4i __attribute__((vector_size(16)));   // matches async-LDS builtin param type

#define B_  8
#define N_  2048
#define M_  512
#define K_  128
#define R2_ 0.0625f
#define BN_ 1024
#define SLOPE_ 0.2f
#define FINF (__builtin_inff())

// CDNA5 async global->LDS copy path (guarded; falls back to uint4 copy).
#if defined(__has_builtin)
#if __has_builtin(__builtin_amdgcn_global_load_async_to_lds_b128) && \
    __has_builtin(__builtin_amdgcn_s_wait_asynccnt)
#define USE_ASYNC_LDS 1
#endif
#endif
#ifndef USE_ASYNC_LDS
#define USE_ASYNC_LDS 0
#endif

// ---------------- helpers ----------------
__device__ __forceinline__ float leaky(float v) { return v > 0.f ? v : SLOPE_ * v; }

__device__ __forceinline__ v8f wmma16(v16h a, v16h b, v8f c) {
  return __builtin_amdgcn_wmma_f32_16x16x32_f16(false, a, false, b, (short)0, c, false, false);
}

// Cooperative global -> LDS copy of `bytes` (multiple of 16, 16B-aligned).
__device__ __forceinline__ void copy_g2l(const void* __restrict__ g, void* l,
                                         int bytes, int tid, int nthreads) {
  int n16 = bytes >> 4;
#if USE_ASYNC_LDS
  g_v4i* gs = (g_v4i*)g;
  g_v4i* ls = (g_v4i*)l;
  for (int i = tid; i < n16; i += nthreads) {
    __builtin_amdgcn_global_load_async_to_lds_b128(gs + i, ls + i, 0, 0);
  }
#else
  const uint4* gs = (const uint4*)g;
  uint4* ls = (uint4*)l;
  for (int i = tid; i < n16; i += nthreads) ls[i] = gs[i];
#endif
}

__device__ __forceinline__ void copy_g2l_wait() {
#if USE_ASYNC_LDS
  __builtin_amdgcn_s_wait_asynccnt(0);
#endif
}

// A fragment (16x32 f16) from row-major f16 array with pitch (halves).
// Layout (ISA 7.12.2): lane L row = L&15; hi=L>=16; half pairs at K offsets
// {0,2,4,6,16,18,20,22} + hi*8 relative to kBase.
__device__ __forceinline__ v16h ldA(const _Float16* base, int rowBase, int pitch,
                                    int kBase, int lane) {
  int r  = rowBase + (lane & 15);
  int hi = (lane >> 4) & 1;
  const unsigned* p32 = (const unsigned*)(base + (size_t)r * pitch + kBase + hi * 8);
  union { v16h v; unsigned u[8]; } f;
  f.u[0] = p32[0];  f.u[1] = p32[1];  f.u[2] = p32[2];  f.u[3] = p32[3];
  f.u[4] = p32[8];  f.u[5] = p32[9];  f.u[6] = p32[10]; f.u[7] = p32[11];
  return f.v;
}

// B fragment (32x16 f16) from pre-swizzled weights: 16 contiguous halves per lane.
__device__ __forceinline__ v16h ldB(const _Float16* wsw, int ktiles, int kt, int nt, int lane) {
  const uint4* p4 = (const uint4*)(wsw + ((size_t)(nt * ktiles + kt) * 32 + lane) * 16);
  union { v16h v; uint4 q[2]; } f;
  f.q[0] = p4[0]; f.q[1] = p4[1];
  return f.v;
}

// Store C tile (f32) with bias + leaky into row-major f16 buffer.
__device__ __forceinline__ void stC(_Float16* dst, int pitch, int rowBase, int colBase,
                                    v8f c, const float* bias, int lane) {
  int n  = colBase + (lane & 15);
  int hi = (lane >> 4) & 1;
  float bb = bias[n];
#pragma unroll
  for (int v = 0; v < 8; ++v)
    dst[(size_t)(rowBase + hi * 8 + v) * pitch + n] = (_Float16)leaky(c[v] + bb);
}

// ---------------- prep kernels ----------------
__global__ void cvt_x_kernel(const float* __restrict__ x, _Float16* __restrict__ xh, int n) {
  int i = blockIdx.x * 256 + threadIdx.x;
  if (i < n) xh[i] = (_Float16)x[i];
}

// Swizzle W [Kdim x Ndim] row-major -> B-fragment tiles, pad K with zeros.
__global__ void swz_kernel(const float* __restrict__ W, _Float16* __restrict__ out,
                           int Kdim, int Ndim, int ktiles, int total) {
  int t = blockIdx.x * 256 + threadIdx.x;
  if (t >= total) return;
  int j    = t & 15;
  int lane = (t >> 4) & 31;
  int tile = t >> 9;
  int kt   = tile % ktiles;
  int nt   = tile / ktiles;
  int hi   = lane >> 4;
  int n    = nt * 16 + (lane & 15);
  int k    = kt * 32 + (j & 7) + (j >> 3) * 16 + hi * 8;
  out[t] = (k < Kdim) ? (_Float16)W[(size_t)k * Ndim + n] : (_Float16)0.f;
}

// ---------------- FPS ----------------
__global__ __launch_bounds__(256) void fps_kernel(const float* __restrict__ pos,
                                                  int* __restrict__ idx,
                                                  float* __restrict__ centers) {
  int b = blockIdx.x, tid = threadIdx.x;
  __shared__ float px[N_], py[N_], pz[N_];
  __shared__ float rv[256];
  __shared__ int   ri[256];
  for (int i = tid; i < N_; i += 256) {
    px[i] = pos[(size_t)(b * N_ + i) * 3 + 0];
    py[i] = pos[(size_t)(b * N_ + i) * 3 + 1];
    pz[i] = pos[(size_t)(b * N_ + i) * 3 + 2];
  }
  __syncthreads();
  float d[8];
#pragma unroll
  for (int j = 0; j < 8; ++j) d[j] = FINF;
  int last = 0;
  for (int m = 0; m < M_; ++m) {
    if (tid == 0) {
      idx[b * M_ + m] = last;
      centers[(size_t)(b * M_ + m) * 3 + 0] = px[last];
      centers[(size_t)(b * M_ + m) * 3 + 1] = py[last];
      centers[(size_t)(b * M_ + m) * 3 + 2] = pz[last];
    }
    float cx = px[last], cy = py[last], cz = pz[last];
    float bestv = -1.f; int besti = tid * 8;
#pragma unroll
    for (int j = 0; j < 8; ++j) {
      int i = tid * 8 + j;
      float dx = px[i] - cx, dy = py[i] - cy, dz = pz[i] - cz;
      float dd = dx * dx + dy * dy + dz * dz;
      d[j] = fminf(d[j], dd);
      if (d[j] > bestv) { bestv = d[j]; besti = i; }   // first-max within thread
    }
    rv[tid] = bestv; ri[tid] = besti;
    __syncthreads();
    for (int s = 128; s > 0; s >>= 1) {
      if (tid < s) {
        float v2 = rv[tid + s]; int i2 = ri[tid + s];
        if (v2 > rv[tid] || (v2 == rv[tid] && i2 < ri[tid])) { rv[tid] = v2; ri[tid] = i2; }
      }
      __syncthreads();
    }
    last = ri[0];
    __syncthreads();
  }
}

// ---------------- ball query (bitonic top-K) ----------------
__global__ __launch_bounds__(256) void ballq_kernel(const float* __restrict__ pos,
                                                    const float* __restrict__ centers,
                                                    int* __restrict__ nbr,
                                                    unsigned* __restrict__ validw,
                                                    float* __restrict__ out_xidx,
                                                    float* __restrict__ out_yidx) {
  int bm = blockIdx.x, b = bm / M_, m = bm % M_;
  __shared__ float kd[N_];
  __shared__ int   ki[N_];
  float cx = centers[(size_t)bm * 3 + 0];
  float cy = centers[(size_t)bm * 3 + 1];
  float cz = centers[(size_t)bm * 3 + 2];
  for (int i = threadIdx.x; i < N_; i += 256) {
    float dx = pos[(size_t)(b * N_ + i) * 3 + 0] - cx;
    float dy = pos[(size_t)(b * N_ + i) * 3 + 1] - cy;
    float dz = pos[(size_t)(b * N_ + i) * 3 + 2] - cz;
    float dd = dx * dx + dy * dy + dz * dz;
    kd[i] = (dd <= R2_) ? dd : FINF;
    ki[i] = i;
  }
  __syncthreads();
  for (int k2 = 2; k2 <= N_; k2 <<= 1) {
    for (int j = k2 >> 1; j > 0; j >>= 1) {
      for (int i = threadIdx.x; i < N_; i += 256) {
        int ixj = i ^ j;
        if (ixj > i) {
          float da = kd[i], db = kd[ixj];
          int   ia = ki[i], ib = ki[ixj];
          bool up = ((i & k2) == 0);
          bool gt = (da > db) || (da == db && ia > ib);
          if (gt == up) { kd[i] = db; kd[ixj] = da; ki[i] = ib; ki[ixj] = ia; }
        }
      }
      __syncthreads();
    }
  }
  for (int k = threadIdx.x; k < K_; k += 256) {
    int n = ki[k];
    bool val = (kd[k] < FINF);
    nbr[(size_t)bm * K_ + k]      = n;
    out_xidx[(size_t)bm * K_ + k] = val ? (float)(n + b * N_) : -1.f;
    out_yidx[(size_t)bm * K_ + k] = val ? (float)(m + b * M_) : -1.f;
  }
  if (threadIdx.x < 4) {
    unsigned wv = 0;
    for (int bit = 0; bit < 32; ++bit)
      if (kd[threadIdx.x * 32 + bit] < FINF) wv |= (1u << bit);
    validw[(size_t)bm * 4 + threadIdx.x] = wv;
  }
}

// ---------------- per-center neighbor MLP (WMMA) + masked max ----------------
// LDS layout (halves unless noted):
//   W1l 6144 | W2l 8192 | W3l 65536 | h0 12288 | h1 8192 | h2 16384 | aggW 4096 f32
__global__ __launch_bounds__(256) void mlp_local_kernel(
    const _Float16* __restrict__ xh, const float* __restrict__ pos,
    const float* __restrict__ centers, const int* __restrict__ nbr,
    const unsigned* __restrict__ validw,
    const _Float16* __restrict__ W1sw, const float* __restrict__ b1,
    const _Float16* __restrict__ W2sw, const float* __restrict__ b2,
    const _Float16* __restrict__ W3sw, const float* __restrict__ b3,
    _Float16* __restrict__ g0) {
  extern __shared__ char smem[];
  _Float16* W1l  = (_Float16*)smem;            //  3*4*512
  _Float16* W2l  = W1l + 6144;                 //  2*8*512
  _Float16* W3l  = W2l + 8192;                 //  4*32*512
  _Float16* h0   = W3l + 65536;                // 128 x 96
  _Float16* h1   = h0 + 128 * 96;              // 128 x 64
  _Float16* h2   = h1 + 128 * 64;              // 128 x 128
  float*    aggW = (float*)(h2 + 128 * 128);   // 8 x 512

  int bm = blockIdx.x, b = bm / M_;
  int tid = threadIdx.x, lane = tid & 31, w = tid >> 5;
  float cx = centers[(size_t)bm * 3 + 0];
  float cy = centers[(size_t)bm * 3 + 1];
  float cz = centers[(size_t)bm * 3 + 2];

  // stage all weights into LDS (async if available)
  copy_g2l(W1sw, W1l, 6144 * 2, tid, 256);
  copy_g2l(W2sw, W2l, 8192 * 2, tid, 256);
  copy_g2l(W3sw, W3l, 65536 * 2, tid, 256);

  { // gather xj | rel | pad into h0 (f16)
    int r = tid >> 1, halfsel = tid & 1;
    int nidx = nbr[(size_t)bm * K_ + r];
    _Float16* row = h0 + (size_t)r * 96;
    const _Float16* src = xh + (size_t)(b * N_ + nidx) * 64;
    if (halfsel == 0) {
      for (int c = 0; c < 48; ++c) row[c] = src[c];
    } else {
      for (int c = 48; c < 64; ++c) row[c] = src[c];
      const float* pp = pos + (size_t)(b * N_ + nidx) * 3;
      row[64] = (_Float16)(pp[0] - cx);
      row[65] = (_Float16)(pp[1] - cy);
      row[66] = (_Float16)(pp[2] - cz);
      for (int c = 67; c < 96; ++c) row[c] = (_Float16)0.f;
    }
  }
  copy_g2l_wait();
  __syncthreads();

  int rowBase = w * 16;
  // layer1: [16x96]x[96x64]
  v16h a0 = ldA(h0, rowBase, 96, 0, lane);
  v16h a1 = ldA(h0, rowBase, 96, 32, lane);
  v16h a2 = ldA(h0, rowBase, 96, 64, lane);
#pragma unroll
  for (int nt = 0; nt < 4; ++nt) {
    v8f c = {};
    c = wmma16(a0, ldB(W1l, 3, 0, nt, lane), c);
    c = wmma16(a1, ldB(W1l, 3, 1, nt, lane), c);
    c = wmma16(a2, ldB(W1l, 3, 2, nt, lane), c);
    stC(h1, 64, rowBase, nt * 16, c, b1, lane);
  }
  __syncthreads();

  // layer2: [16x64]x[64x128]
  v16h c0 = ldA(h1, rowBase, 64, 0, lane);
  v16h c1 = ldA(h1, rowBase, 64, 32, lane);
#pragma unroll
  for (int nt = 0; nt < 8; ++nt) {
    v8f c = {};
    c = wmma16(c0, ldB(W2l, 2, 0, nt, lane), c);
    c = wmma16(c1, ldB(W2l, 2, 1, nt, lane), c);
    stC(h2, 128, rowBase, nt * 16, c, b2, lane);
  }
  __syncthreads();

  // layer3: [16x128]x[128x512] + masked row-max
  v16h d0 = ldA(h2, rowBase, 128, 0, lane);
  v16h d1 = ldA(h2, rowBase, 128, 32, lane);
  v16h d2 = ldA(h2, rowBase, 128, 64, lane);
  v16h d3 = ldA(h2, rowBase, 128, 96, lane);
  unsigned vb;
  {
    int hi = (lane >> 4) & 1;
    int rstart = rowBase + hi * 8;
    unsigned word = validw[(size_t)bm * 4 + (rstart >> 5)];
    vb = (word >> (rstart & 31)) & 0xffu;
  }
  for (int nt = 0; nt < 32; ++nt) {
    v8f c = {};
    c = wmma16(d0, ldB(W3l, 4, 0, nt, lane), c);
    c = wmma16(d1, ldB(W3l, 4, 1, nt, lane), c);
    c = wmma16(d2, ldB(W3l, 4, 2, nt, lane), c);
    c = wmma16(d3, ldB(W3l, 4, 3, nt, lane), c);
    float bb = b3[nt * 16 + (lane & 15)];
    float mx = -FINF;
#pragma unroll
    for (int v = 0; v < 8; ++v) {
      float val = leaky(c[v] + bb);
      if ((vb >> v) & 1u) mx = fmaxf(mx, val);
    }
    mx = fmaxf(mx, __shfl_xor(mx, 16, 32));
    if (lane < 16) aggW[(size_t)w * 512 + nt * 16 + lane] = mx;
  }
  __syncthreads();

  for (int col = tid; col < 512; col += 256) {
    float mx = aggW[col];
#pragma unroll
    for (int ww = 1; ww < 8; ++ww) mx = fmaxf(mx, aggW[(size_t)ww * 512 + col]);
    g0[(size_t)bm * 544 + col] = (_Float16)mx;
  }
  if (tid < 32) {
    float v = (tid == 0) ? cx : (tid == 1) ? cy : (tid == 2) ? cz : 0.f;
    g0[(size_t)bm * 544 + 512 + tid] = (_Float16)v;
  }
}

// ---------------- g0 x W4 -> g1 ----------------
__global__ __launch_bounds__(256) void mlp_g1_kernel(const _Float16* __restrict__ g0,
                                                     const _Float16* __restrict__ W4sw,
                                                     const float* __restrict__ b4,
                                                     _Float16* __restrict__ g1) {
  extern __shared__ char smem[];
  _Float16* Al = (_Float16*)smem;              // 128 x 544
  int tid = threadIdx.x, lane = tid & 31, w = tid >> 5;
  int rBase = blockIdx.x * 128 + w * 16;
  copy_g2l(g0 + (size_t)blockIdx.x * 128 * 544, Al, 128 * 544 * 2, tid, 256);
  copy_g2l_wait();
  __syncthreads();
  for (int ch = 0; ch < 8; ++ch) {
    v8f c[4] = {{}, {}, {}, {}};
    for (int kt = 0; kt < 17; ++kt) {
      v16h a = ldA(Al, w * 16, 544, kt * 32, lane);
#pragma unroll
      for (int q = 0; q < 4; ++q)
        c[q] = wmma16(a, ldB(W4sw, 17, kt, ch * 4 + q, lane), c[q]);
    }
    int hi = (lane >> 4) & 1;
#pragma unroll
    for (int q = 0; q < 4; ++q) {
      int n = (ch * 4 + q) * 16 + (lane & 15);
      float bb = b4[n];
#pragma unroll
      for (int v = 0; v < 8; ++v)
        g1[(size_t)(rBase + v + hi * 8) * 512 + n] = (_Float16)leaky(c[q][v] + bb);
    }
  }
}

// ---------------- g1 x W5 -> mean/std ----------------
__global__ __launch_bounds__(256) void mlp_out_kernel(const _Float16* __restrict__ g1,
                                                      const _Float16* __restrict__ W5sw,
                                                      const float* __restrict__ b5,
                                                      float* __restrict__ out) {
  extern __shared__ char smem[];
  _Float16* Al = (_Float16*)smem;              // 128 x 512
  const size_t OFF_STD = (size_t)B_ * M_ * BN_;
  int tid = threadIdx.x, lane = tid & 31, w = tid >> 5;
  int rBase = blockIdx.x * 128 + w * 16;
  copy_g2l(g1 + (size_t)blockIdx.x * 128 * 512, Al, 128 * 512 * 2, tid, 256);
  copy_g2l_wait();
  __syncthreads();
  for (int ch = 0; ch < 32; ++ch) {
    v8f c[4] = {{}, {}, {}, {}};
    for (int kt = 0; kt < 16; ++kt) {
      v16h a = ldA(Al, w * 16, 512, kt * 32, lane);
#pragma unroll
      for (int q = 0; q < 4; ++q)
        c[q] = wmma16(a, ldB(W5sw, 16, kt, ch * 4 + q, lane), c[q]);
    }
    int hi = (lane >> 4) & 1;
#pragma unroll
    for (int q = 0; q < 4; ++q) {
      int n = (ch * 4 + q) * 16 + (lane & 15);
      float bb = b5[n];
      if (n < BN_) {
#pragma unroll
        for (int v = 0; v < 8; ++v)
          out[(size_t)(rBase + v + hi * 8) * BN_ + n] = c[q][v] + bb;
      } else {
#pragma unroll
        for (int v = 0; v < 8; ++v)
          out[OFF_STD + (size_t)(rBase + v + hi * 8) * BN_ + (n - BN_)] =
              expf(0.5f * (c[q][v] + bb));
      }
    }
  }
}

// ---------------- launch ----------------
extern "C" void kernel_launch(void* const* d_in, const int* in_sizes, int n_in,
                              void* d_out, int out_size, void* d_ws, size_t ws_size,
                              hipStream_t stream) {
  (void)in_sizes; (void)n_in; (void)out_size; (void)ws_size;
  const float* x   = (const float*)d_in[0];
  const float* pos = (const float*)d_in[1];
  const float* W1  = (const float*)d_in[3];
  const float* b1  = (const float*)d_in[4];
  const float* W2  = (const float*)d_in[5];
  const float* b2  = (const float*)d_in[6];
  const float* W3  = (const float*)d_in[7];
  const float* b3  = (const float*)d_in[8];
  const float* W4  = (const float*)d_in[9];
  const float* b4  = (const float*)d_in[10];
  const float* W5  = (const float*)d_in[11];
  const float* b5  = (const float*)d_in[12];
  float* out = (float*)d_out;

  const size_t BM = (size_t)B_ * M_;
  char* ws = (char*)d_ws;
  size_t off = 0;
  auto take = [&](size_t bytes) { size_t r = off; off += (bytes + 255) & ~(size_t)255; return r; };
  int*      w_idx   = (int*)(ws + take(BM * 4));
  float*    w_cent  = (float*)(ws + take(BM * 3 * 4));
  int*      w_nbr   = (int*)(ws + take(BM * K_ * 4));
  unsigned* w_valid = (unsigned*)(ws + take(BM * 4 * 4));
  _Float16* w_xh    = (_Float16*)(ws + take((size_t)B_ * N_ * 64 * 2));
  _Float16* w_W1    = (_Float16*)(ws + take((size_t)3 * 4 * 512 * 2));     //  96x64
  _Float16* w_W2    = (_Float16*)(ws + take((size_t)2 * 8 * 512 * 2));     //  64x128
  _Float16* w_W3    = (_Float16*)(ws + take((size_t)4 * 32 * 512 * 2));    // 128x512
  _Float16* w_W4    = (_Float16*)(ws + take((size_t)17 * 32 * 512 * 2));   // 544x512
  _Float16* w_W5    = (_Float16*)(ws + take((size_t)16 * 128 * 512 * 2));  // 512x2048
  _Float16* w_g0    = (_Float16*)(ws + take(BM * 544 * 2));
  _Float16* w_g1    = (_Float16*)(ws + take(BM * 512 * 2));

  const size_t OFF_XIDX = 2 * BM * BN_;
  const size_t OFF_YIDX = OFF_XIDX + BM * K_;

  int nx = B_ * N_ * 64;
  cvt_x_kernel<<<(nx + 255) / 256, 256, 0, stream>>>(x, w_xh, nx);
  swz_kernel<<<(3 * 4 * 512 + 255) / 256, 256, 0, stream>>>(W1, w_W1, 67, 64, 3, 3 * 4 * 512);
  swz_kernel<<<(2 * 8 * 512 + 255) / 256, 256, 0, stream>>>(W2, w_W2, 64, 128, 2, 2 * 8 * 512);
  swz_kernel<<<(4 * 32 * 512 + 255) / 256, 256, 0, stream>>>(W3, w_W3, 128, 512, 4, 4 * 32 * 512);
  swz_kernel<<<(17 * 32 * 512 + 255) / 256, 256, 0, stream>>>(W4, w_W4, 515, 512, 17, 17 * 32 * 512);
  swz_kernel<<<(16 * 128 * 512 + 255) / 256, 256, 0, stream>>>(W5, w_W5, 512, 2048, 16, 16 * 128 * 512);

  fps_kernel<<<B_, 256, 0, stream>>>(pos, w_idx, w_cent);
  ballq_kernel<<<B_ * M_, 256, 0, stream>>>(pos, w_cent, w_nbr, w_valid,
                                            out + OFF_XIDX, out + OFF_YIDX);
  // LDS: weights (6144+8192+65536)*2 + h buffers (12288+8192+16384)*2 + 8*512*4
  size_t smem3 = (size_t)(6144 + 8192 + 65536 + 12288 + 8192 + 16384) * 2 + (size_t)8 * 512 * 4;
  mlp_local_kernel<<<B_ * M_, 256, smem3, stream>>>(w_xh, pos, w_cent, w_nbr, w_valid,
                                                    w_W1, b1, w_W2, b2, w_W3, b3, w_g0);
  mlp_g1_kernel<<<(int)(BM / 128), 256, (size_t)128 * 544 * 2, stream>>>(w_g0, w_W4, b4, w_g1);
  mlp_out_kernel<<<(int)(BM / 128), 256, (size_t)128 * 512 * 2, stream>>>(w_g1, w_W5, b5, out);
}
